// FlashAttention_35029753266717
// MI455X (gfx1250) — matrix operand order repaired and checked by reading, not compiled, not run on hardware
//
#include <hip/hip_runtime.h>
#include <hip/hip_bf16.h>

// ---------------------------------------------------------------------------
// MHA block for MI455X (gfx1250): all matmuls through v_wmma_f32_16x16x32_f16.
//   B=4, L=2048, D=1024, H=16, DH=64, SCALE = 1/8
// Flash attention stages K/V tiles into LDS per block (8 waves share) via the
// CDNA5 async-to-LDS path (ASYNCcnt) with double buffering.
// ---------------------------------------------------------------------------

typedef __attribute__((ext_vector_type(16))) _Float16 v16h;
typedef __attribute__((ext_vector_type(8)))  float    v8f;

#define BB 4
#define LL 2048
#define DD 1024
#define HH 16
#define DH 64
#define SCALE 0.125f

union H2U { unsigned u; _Float16 h[2]; };

static __device__ __forceinline__ v8f wmma16(v16h a, v16h b, v8f c) {
    return __builtin_amdgcn_wmma_f32_16x16x32_f16(
        /*neg_a=*/false, a, /*neg_b=*/false, b,
        /*c_mod=*/(short)0, c, /*reuse_a=*/false, /*reuse_b=*/false);
}

// A fragment 16x32 f16: lane (g = lane>>4, m = lane&15) holds row m.
// VGPR v (<4): K = 8g + 2v + h ; (>=4): K = 16 + 8g + 2(v-4) + h.
static __device__ __forceinline__ int a_kpair(int g, int v) {
    return (v < 4) ? (8 * g + 2 * v) : (16 + 8 * g + 2 * (v - 4));
}
// B fragment 32x16 f16: lane holds column n = lane&15; same K map as A.
static __device__ __forceinline__ int b_kpair(int g, int v) {
    return (v < 4) ? (8 * g + 2 * v) : (16 + 8 * g + 2 * (v - 4));
}

// ---------------- async copy to LDS (gfx1250 ASYNCcnt path) ----------------
#if __has_builtin(__builtin_amdgcn_global_load_async_to_lds_b128)
#define HAVE_ASYNC_LDS 1
typedef int v4i_vs __attribute__((vector_size(4 * sizeof(int))));
typedef __attribute__((address_space(1))) v4i_vs* gp1_t;   // global int4*
typedef __attribute__((address_space(3))) v4i_vs* lp3_t;   // LDS int4*
#else
#define HAVE_ASYNC_LDS 0
#endif

static __device__ __forceinline__ void cp_b128_to_lds(_Float16* dst,
                                                      const _Float16* src) {
#if HAVE_ASYNC_LDS
    __builtin_amdgcn_global_load_async_to_lds_b128(
        (gp1_t)(void*)const_cast<_Float16*>(src), (lp3_t)(void*)dst, 0, 0);
#else
    *reinterpret_cast<uint4*>(dst) = *reinterpret_cast<const uint4*>(src);
#endif
}

static __device__ __forceinline__ void wait_async_le2() {
#if HAVE_ASYNC_LDS
#if __has_builtin(__builtin_amdgcn_s_wait_asynccnt)
    __builtin_amdgcn_s_wait_asynccnt(2);
#else
    asm volatile("s_wait_asynccnt 0x2" ::: "memory");
#endif
#endif
}
static __device__ __forceinline__ void wait_async_le0() {
#if HAVE_ASYNC_LDS
#if __has_builtin(__builtin_amdgcn_s_wait_asynccnt)
    __builtin_amdgcn_s_wait_asynccnt(0);
#else
    asm volatile("s_wait_asynccnt 0x0" ::: "memory");
#endif
#endif
}

// ---------------------------------------------------------------------------
// Kernel 0: convert the four weight matrices fp32 -> f16 (row-major kept).
// ---------------------------------------------------------------------------
__global__ void cvt_weights_kernel(const float* __restrict__ wq,
                                   const float* __restrict__ wk,
                                   const float* __restrict__ wv,
                                   const float* __restrict__ wo,
                                   _Float16* __restrict__ w16) {
    const unsigned tid = blockIdx.x * blockDim.x + threadIdx.x;   // 4 * 2^20
    const unsigned arr = tid >> 20;
    const unsigned rem = tid & ((1u << 20) - 1u);
    const float* src = (arr == 0) ? wq : (arr == 1) ? wk : (arr == 2) ? wv : wo;
    w16[(size_t)arr * (1u << 20) + rem] = (_Float16)src[rem];
}

// ---------------------------------------------------------------------------
// Kernel 1: fused QKV projection.  y = x @ W^T + b, scattered to head layout.
//   Wave computes 32(M) x 64(N); K loop of 32 -> 8 WMMA per step.
//   Q,K out: [B][H][L][DH] f16 ; V out transposed: [B][H][DH][L] f16.
// ---------------------------------------------------------------------------
__global__ void __launch_bounds__(256)
qkv_proj_kernel(const float* __restrict__ xq, const float* __restrict__ xk,
                const float* __restrict__ xv,
                const _Float16* __restrict__ w16,   // 4 matrices packed
                const float* __restrict__ bq, const float* __restrict__ bk,
                const float* __restrict__ bv,
                _Float16* __restrict__ outQ, _Float16* __restrict__ outK,
                _Float16* __restrict__ outVT) {
    const int lane = threadIdx.x & 31;
    const int wave = threadIdx.x >> 5;
    const int g    = lane >> 4;
    const int ln   = lane & 15;
    const int p    = blockIdx.z;

    const float*    X    = (p == 0) ? xq : (p == 1) ? xk : xv;
    const _Float16* W    = w16 + (size_t)p * (DD * DD);
    const float*    bias = (p == 0) ? bq : (p == 1) ? bk : bv;

    const int m0 = (blockIdx.y * 8 + wave) * 32;     // row tile in [0,8192)
    const int n0 = blockIdx.x * 64;                  // col tile in [0,1024)

    v8f acc[8];                                      // [mi*4 + nt]
#pragma unroll
    for (int i = 0; i < 8; ++i) acc[i] = (v8f){0,0,0,0,0,0,0,0};

    const float* xrow0 = X + (size_t)(m0 + ln) * DD;
    const float* xrow1 = X + (size_t)(m0 + 16 + ln) * DD;

    for (int k0 = 0; k0 < DD; k0 += 32) {
        v16h af0, af1;
#pragma unroll
        for (int v = 0; v < 8; ++v) {
            const int kk = k0 + a_kpair(g, v);
            af0[2 * v]     = (_Float16)xrow0[kk];
            af0[2 * v + 1] = (_Float16)xrow0[kk + 1];
            af1[2 * v]     = (_Float16)xrow1[kk];
            af1[2 * v + 1] = (_Float16)xrow1[kk + 1];
        }
#pragma unroll
        for (int nt = 0; nt < 4; ++nt) {
            const int j = n0 + nt * 16 + ln;          // W row (since B = W^T)
            v16h bf;
#pragma unroll
            for (int v = 0; v < 8; ++v) {
                H2U t;
                t.u = *reinterpret_cast<const unsigned*>(
                    &W[(size_t)j * DD + k0 + b_kpair(g, v)]);
                bf[2 * v]     = t.h[0];
                bf[2 * v + 1] = t.h[1];
            }
            acc[nt]     = wmma16(af0, bf, acc[nt]);
            acc[4 + nt] = wmma16(af1, bf, acc[4 + nt]);
        }
    }

#pragma unroll
    for (int nt = 0; nt < 4; ++nt) {
        const int j  = n0 + nt * 16 + ln;
        const int h  = j >> 6;
        const int dh = j & 63;
        const float bj = bias[j];
#pragma unroll
        for (int mi = 0; mi < 2; ++mi) {
#pragma unroll
            for (int r = 0; r < 8; ++r) {
                const int mRow = m0 + mi * 16 + 8 * g + r;
                const int b    = mRow >> 11;
                const int l    = mRow & (LL - 1);
                const _Float16 val = (_Float16)(acc[mi * 4 + nt][r] + bj);
                if (p == 2) {
                    outVT[(((size_t)b * HH + h) * DH + dh) * LL + l] = val;
                } else {
                    _Float16* dst = (p == 0) ? outQ : outK;
                    dst[(((size_t)b * HH + h) * LL + l) * DH + dh] = val;
                }
            }
        }
    }
}

// ---------------------------------------------------------------------------
// Kernel 2: flash attention.  Block = 8 waves sharing one (b,h); each wave
//   owns 16 q rows.  K/V tiles (32 kv positions) staged in LDS per block,
//   double-buffered via async-to-LDS.  Per KV step: 4 WMMA scores + online
//   softmax (shfl_xor width-16 row reductions) + P via LDS re-layout + 4 WMMA.
// ---------------------------------------------------------------------------
__global__ void __launch_bounds__(256)
flash_attn_kernel(const _Float16* __restrict__ Q, const _Float16* __restrict__ K,
                  const _Float16* __restrict__ VT, _Float16* __restrict__ AO) {
    __shared__ _Float16 ktile[2][32 * 64];           // [lk][dh]
    __shared__ _Float16 vtile[2][64 * 32];           // [dh][lk]
    __shared__ _Float16 ptile[8][16 * 32];           // per-wave P staging

    const int tid  = threadIdx.x;
    const int lane = tid & 31;
    const int wave = tid >> 5;
    const int g    = lane >> 4;
    const int ln   = lane & 15;

    const int b  = blockIdx.z;
    const int h  = blockIdx.y;
    const int q0 = blockIdx.x * 128 + wave * 16;

    const _Float16* Qbh = Q  + ((size_t)b * HH + h) * LL * DH;
    const _Float16* Kbh = K  + ((size_t)b * HH + h) * LL * DH;
    const _Float16* Vbh = VT + ((size_t)b * HH + h) * DH * LL;
    _Float16* lds = &ptile[wave][0];

    // Cooperative tile loader: each of 256 threads moves one b128 of K and V.
    const int krow = tid >> 3, kcol = (tid & 7) * 8;     // K tile 32x64
    const int vrow = tid >> 2, vcol = (tid & 3) * 8;     // V tile 64x32
    const _Float16* ksrc = Kbh + (size_t)krow * DH + kcol;
    const _Float16* vsrc = Vbh + (size_t)vrow * LL + vcol;

    // Q fragments held in registers for the whole KV loop (dh 0-31, 32-63).
    v16h qf0, qf1;
    {
        const _Float16* qrow = Qbh + (size_t)(q0 + ln) * DH;
#pragma unroll
        for (int v = 0; v < 8; ++v) {
            const int kk = a_kpair(g, v);
            H2U t0, t1;
            t0.u = *reinterpret_cast<const unsigned*>(&qrow[kk]);
            t1.u = *reinterpret_cast<const unsigned*>(&qrow[32 + kk]);
            qf0[2 * v] = t0.h[0]; qf0[2 * v + 1] = t0.h[1];
            qf1[2 * v] = t1.h[0]; qf1[2 * v + 1] = t1.h[1];
        }
    }

    const v8f vzero = {0,0,0,0,0,0,0,0};
    v8f o0 = vzero, o1 = vzero, o2 = vzero, o3 = vzero;
    float mrow[8], lrow[8];
#pragma unroll
    for (int r = 0; r < 8; ++r) { mrow[r] = -3.0e38f; lrow[r] = 0.0f; }

    const int NT = LL / 32;
    // Prologue: issue tile 0 into buffer 0.
    cp_b128_to_lds(&ktile[0][tid * 8], ksrc);
    cp_b128_to_lds(&vtile[0][tid * 8], vsrc);

    for (int it = 0; it < NT; ++it) {
        const int buf = it & 1;
        // Issue next tile into the other buffer.
        if (it + 1 < NT) {
            const int kv1 = (it + 1) * 32;
            cp_b128_to_lds(&ktile[buf ^ 1][tid * 8], ksrc + (size_t)kv1 * DH);
            cp_b128_to_lds(&vtile[buf ^ 1][tid * 8], vsrc + kv1);
            wait_async_le2();                         // tile `it` complete
        } else {
            wait_async_le0();
        }
        __syncthreads();                              // publish tile `it`

        const _Float16* kb = &ktile[buf][0];
        const _Float16* vb = &vtile[buf][0];

        // ---- scores: two 16x16 subtiles, each K=64 (2 WMMA) ----
        v8f s[2];
#pragma unroll
        for (int sub = 0; sub < 2; ++sub) {
            const _Float16* krow16 = kb + (sub * 16 + ln) * 64;
            v16h bf0, bf1;
#pragma unroll
            for (int v = 0; v < 8; ++v) {
                const int kk = b_kpair(g, v);
                H2U t0, t1;
                t0.u = *reinterpret_cast<const unsigned*>(&krow16[kk]);
                t1.u = *reinterpret_cast<const unsigned*>(&krow16[32 + kk]);
                bf0[2 * v] = t0.h[0]; bf0[2 * v + 1] = t0.h[1];
                bf1[2 * v] = t1.h[0]; bf1[2 * v + 1] = t1.h[1];
            }
            v8f c = vzero;
            c = wmma16(qf0, bf0, c);
            c = wmma16(qf1, bf1, c);
            s[sub] = c;
        }

        // ---- online softmax per row; P(f16) staged into per-wave LDS ----
        float alpha[8];
#pragma unroll
        for (int r = 0; r < 8; ++r) {
            const float s0 = s[0][r] * SCALE;
            const float s1 = s[1][r] * SCALE;
            float mx = fmaxf(s0, s1);
            mx = fmaxf(mx, __shfl_xor(mx, 1, 16));
            mx = fmaxf(mx, __shfl_xor(mx, 2, 16));
            mx = fmaxf(mx, __shfl_xor(mx, 4, 16));
            mx = fmaxf(mx, __shfl_xor(mx, 8, 16));
            const float mnew = fmaxf(mrow[r], mx);
            const float p0 = __expf(s0 - mnew);
            const float p1 = __expf(s1 - mnew);
            float ps = p0 + p1;
            ps += __shfl_xor(ps, 1, 16);
            ps += __shfl_xor(ps, 2, 16);
            ps += __shfl_xor(ps, 4, 16);
            ps += __shfl_xor(ps, 8, 16);
            alpha[r] = __expf(mrow[r] - mnew);
            lrow[r]  = lrow[r] * alpha[r] + ps;
            mrow[r]  = mnew;
            const int row = 8 * g + r;                // C layout row
            lds[row * 32 + ln]      = (_Float16)p0;   // cols 0..15
            lds[row * 32 + 16 + ln] = (_Float16)p1;   // cols 16..31
        }
#pragma unroll
        for (int r = 0; r < 8; ++r) {
            o0[r] *= alpha[r]; o1[r] *= alpha[r];
            o2[r] *= alpha[r]; o3[r] *= alpha[r];
        }

        // ---- re-load P as A fragment (DS ops in-order within a wave) ----
        v16h pf;
#pragma unroll
        for (int v = 0; v < 8; ++v) {
            H2U t;
            t.u = *reinterpret_cast<const unsigned*>(&lds[ln * 32 + a_kpair(g, v)]);
            pf[2 * v] = t.h[0]; pf[2 * v + 1] = t.h[1];
        }

        // ---- O += P @ V : 4 WMMA across DH ----
        v8f* os[4] = {&o0, &o1, &o2, &o3};
#pragma unroll
        for (int nt = 0; nt < 4; ++nt) {
            const _Float16* vrow16 = vb + (nt * 16 + ln) * 32;
            v16h vf;
#pragma unroll
            for (int v = 0; v < 8; ++v) {
                H2U t;
                t.u = *reinterpret_cast<const unsigned*>(&vrow16[b_kpair(g, v)]);
                vf[2 * v] = t.h[0]; vf[2 * v + 1] = t.h[1];
            }
            *os[nt] = wmma16(pf, vf, *os[nt]);
        }

        __syncthreads();            // all waves done with tile `it` buffer
    }

    // ---- normalize and store attention output as [B][L][D] f16 ----
    v8f os[4] = {o0, o1, o2, o3};
#pragma unroll
    for (int r = 0; r < 8; ++r) {
        const float inv = 1.0f / lrow[r];
        const int row = q0 + 8 * g + r;
#pragma unroll
        for (int nt = 0; nt < 4; ++nt) {
            const int j = h * DH + nt * 16 + ln;
            AO[((size_t)b * LL + row) * DD + j] = (_Float16)(os[nt][r] * inv);
        }
    }
}

// ---------------------------------------------------------------------------
// Kernel 3: output projection.  out = AO(f16) @ Wo^T + bo -> fp32 d_out.
//   Wave computes 32(M) x 64(N); 8 WMMA per K-step of 32.
// ---------------------------------------------------------------------------
__global__ void __launch_bounds__(256)
out_proj_kernel(const _Float16* __restrict__ A, const _Float16* __restrict__ Wo,
                const float* __restrict__ bo, float* __restrict__ out) {
    const int lane = threadIdx.x & 31;
    const int wave = threadIdx.x >> 5;
    const int g    = lane >> 4;
    const int ln   = lane & 15;

    const int m0 = (blockIdx.y * 8 + wave) * 32;
    const int n0 = blockIdx.x * 64;

    v8f acc[8];
#pragma unroll
    for (int i = 0; i < 8; ++i) acc[i] = (v8f){0,0,0,0,0,0,0,0};

    const _Float16* arow0 = A + (size_t)(m0 + ln) * DD;
    const _Float16* arow1 = A + (size_t)(m0 + 16 + ln) * DD;

    for (int k0 = 0; k0 < DD; k0 += 32) {
        v16h af0, af1;
#pragma unroll
        for (int v = 0; v < 8; ++v) {
            const int kk = k0 + a_kpair(g, v);
            H2U t0, t1;
            t0.u = *reinterpret_cast<const unsigned*>(&arow0[kk]);
            t1.u = *reinterpret_cast<const unsigned*>(&arow1[kk]);
            af0[2 * v] = t0.h[0]; af0[2 * v + 1] = t0.h[1];
            af1[2 * v] = t1.h[0]; af1[2 * v + 1] = t1.h[1];
        }
#pragma unroll
        for (int nt = 0; nt < 4; ++nt) {
            const int j = n0 + nt * 16 + ln;
            v16h bf;
#pragma unroll
            for (int v = 0; v < 8; ++v) {
                H2U t;
                t.u = *reinterpret_cast<const unsigned*>(
                    &Wo[(size_t)j * DD + k0 + b_kpair(g, v)]);
                bf[2 * v] = t.h[0]; bf[2 * v + 1] = t.h[1];
            }
            acc[nt]     = wmma16(af0, bf, acc[nt]);
            acc[4 + nt] = wmma16(af1, bf, acc[4 + nt]);
        }
    }

#pragma unroll
    for (int nt = 0; nt < 4; ++nt) {
        const int j = n0 + nt * 16 + ln;
        const float bj = bo[j];
#pragma unroll
        for (int mi = 0; mi < 2; ++mi) {
#pragma unroll
            for (int r = 0; r < 8; ++r) {
                const int mRow = m0 + mi * 16 + 8 * g + r;
                out[(size_t)mRow * DD + j] = acc[mi * 4 + nt][r] + bj;
            }
        }
    }
}

// ---------------------------------------------------------------------------
// Launcher
// ---------------------------------------------------------------------------
extern "C" void kernel_launch(void* const* d_in, const int* in_sizes, int n_in,
                              void* d_out, int out_size, void* d_ws, size_t ws_size,
                              hipStream_t stream) {
    (void)in_sizes; (void)n_in; (void)out_size; (void)ws_size;

    const float* q  = (const float*)d_in[0];
    const float* k  = (const float*)d_in[1];
    const float* v  = (const float*)d_in[2];
    const float* Wq = (const float*)d_in[3];
    const float* bq = (const float*)d_in[4];
    const float* Wk = (const float*)d_in[5];
    const float* bk = (const float*)d_in[6];
    const float* Wv = (const float*)d_in[7];
    const float* bv = (const float*)d_in[8];
    const float* Wo = (const float*)d_in[9];
    const float* bo = (const float*)d_in[10];
    float* out = (float*)d_out;

    // Workspace layout (f16 elements)
    _Float16* ws  = (_Float16*)d_ws;
    const size_t WN  = (size_t)DD * DD;          // 1M elems per weight matrix
    const size_t TN  = (size_t)BB * LL * DD;     // 8M elems per activation tensor
    _Float16* w16    = ws;                       // 4 weight matrices (q,k,v,o)
    _Float16* q16    = ws + 4 * WN;              // [B][H][L][DH]
    _Float16* k16    = q16 + TN;                 // [B][H][L][DH]
    _Float16* vT16   = k16 + TN;                 // [B][H][DH][L]
    _Float16* ao16   = vT16 + TN;                // [B][L][D]

    // 0) weights fp32 -> f16
    cvt_weights_kernel<<<dim3((4u << 20) / 256), dim3(256), 0, stream>>>(
        Wq, Wk, Wv, Wo, w16);

    // 1) fused QKV projection (grid.z = projection index)
    qkv_proj_kernel<<<dim3(DD / 64, (BB * LL) / (32 * 8), 3), dim3(256), 0, stream>>>(
        q, k, v, w16, bq, bk, bv, q16, k16, vT16);

    // 2) flash attention (block = 8 waves sharing one (b,h))
    flash_attn_kernel<<<dim3(LL / 128, HH, BB), dim3(256), 0, stream>>>(
        q16, k16, vT16, ao16);

    // 3) output projection -> fp32 result
    out_proj_kernel<<<dim3(DD / 64, (BB * LL) / (32 * 8), 1), dim3(256), 0, stream>>>(
        ao16, w16 + 3 * WN, bo, out);
}
